// CausalSelfAttention_1468878815509
// MI455X (gfx1250) — compile-verified, matrix operand order checked by reading, not run on hardware
//
#include <hip/hip_runtime.h>

typedef __attribute__((ext_vector_type(16))) __bf16 v16bf;
typedef __attribute__((ext_vector_type(8)))  __bf16 v8bf;
typedef __attribute__((ext_vector_type(8)))  float  v8f;

#define HIDDEN 1024
#define NHEADS 16
#define HEADD  64
#define SEQ    2048
#define BATCH  2
#define MTOT   (BATCH * SEQ)   // 4096 rows

// ---------------------------------------------------------------------------
// WMMA helper: D = A(16x32 bf16) * B(32x16 bf16) + C(16x16 f32)
// ---------------------------------------------------------------------------
__device__ __forceinline__ v8f wmma_bf16(v16bf a, v16bf b, v8f c) {
  return __builtin_amdgcn_wmma_f32_16x16x32_bf16(
      /*neg_a=*/false, a, /*neg_b=*/false, b,
      /*c_mod=*/(short)0, c, /*reuse_a=*/false, /*reuse_b=*/false);
}

// ---------------------------------------------------------------------------
// Fragment loader per CDNA5 16-bit A/B VGPR striping (CDNA5 ISA):
//   lanes 0-15 : row = lane,    K = {h*8..h*8+7} in v0..v3, {16+h*8..} in v4..v7 (h=0)
//   lanes 16-31: row = lane-16, same with h=1
// Source must be row-major with K contiguous: two 16-byte loads per lane.
// ---------------------------------------------------------------------------
__device__ __forceinline__ v16bf load_frag_k_contig(const __bf16* p, int ld) {
  const int l   = threadIdx.x & 31;
  const int row = l & 15;
  const int h   = l >> 4;
  const __bf16* r = p + (size_t)row * ld + h * 8;
  v8bf lo = *(const v8bf*)(r);        // K = h*8 .. h*8+7
  v8bf hi = *(const v8bf*)(r + 16);   // K = 16+h*8 .. 16+h*8+7
  v16bf out;
#pragma unroll
  for (int i = 0; i < 8; ++i) { out[i] = lo[i]; out[i + 8] = hi[i]; }
  return out;
}

// ---------------------------------------------------------------------------
// fp32 -> bf16 conversions (one-time per launch; ~50 MB total traffic)
// ---------------------------------------------------------------------------
__global__ void cvt_f32_to_bf16(const float* __restrict__ in,
                                __bf16* __restrict__ out, int n) {
  int i = blockIdx.x * blockDim.x + threadIdx.x;
  if (i < n) out[i] = (__bf16)in[i];
}

// w[K][N] fp32  ->  wT[N][K] bf16  (so GEMM B-fragments load contiguously)
__global__ void cvt_tr_f32_to_bf16(const float* __restrict__ w,
                                   __bf16* __restrict__ wT, int K, int N) {
  int idx = blockIdx.x * blockDim.x + threadIdx.x;
  if (idx < K * N) {
    int k = idx / N, n = idx - k * N;
    wT[(size_t)n * K + k] = (__bf16)w[idx];
  }
}

// ---------------------------------------------------------------------------
// GEMM: C[M,N] = A[M,K](bf16) * BT[N,K](bf16)^T + bias[N]
// Block 256 = 8 waves; block tile 128x64; wave tile 32x32 (2x2 WMMA accums).
// mode 0: write f32 to outF.
// mode 1: scatter bf16 into q[B,H,T,D], k[B,H,T,D], v TRANSPOSED [B,H,D,T]
//         (so attention's P@V B-fragments load with contiguous b128).
// ---------------------------------------------------------------------------
__global__ __launch_bounds__(256)
void gemm_bf16_wmma(const __bf16* __restrict__ A, const __bf16* __restrict__ BT,
                    const float* __restrict__ bias, float* __restrict__ outF,
                    __bf16* __restrict__ qO, __bf16* __restrict__ kO,
                    __bf16* __restrict__ vO, int M, int N, int K, int mode) {
  const int wid  = threadIdx.x >> 5;
  const int lane = threadIdx.x & 31;
  const int nloc = lane & 15;
  const int h    = lane >> 4;
  const int m0 = blockIdx.x * 128 + (wid >> 1) * 32;
  const int n0 = blockIdx.y * 64  + (wid & 1)  * 32;

  v8f acc00 = {}, acc01 = {}, acc10 = {}, acc11 = {};
  for (int kt = 0; kt < K; kt += 32) {
    const __bf16* pa0 = A  + (size_t)m0        * K + kt;
    const __bf16* pa1 = A  + (size_t)(m0 + 16) * K + kt;
    const __bf16* pb0 = BT + (size_t)n0        * K + kt;
    const __bf16* pb1 = BT + (size_t)(n0 + 16) * K + kt;
    v16bf a0 = load_frag_k_contig(pa0, K);
    v16bf a1 = load_frag_k_contig(pa1, K);
    v16bf b0 = load_frag_k_contig(pb0, K);
    v16bf b1 = load_frag_k_contig(pb1, K);
    // Prefetch the K-panel 4 steps ahead (global_prefetch_b8)
    __builtin_prefetch(pa0 + 128, 0, 1);
    __builtin_prefetch(pb0 + 128, 0, 1);
    acc00 = wmma_bf16(a0, b0, acc00);
    acc01 = wmma_bf16(a0, b1, acc01);
    acc10 = wmma_bf16(a1, b0, acc10);
    acc11 = wmma_bf16(a1, b1, acc11);
  }

#pragma unroll
  for (int ti = 0; ti < 2; ++ti) {
#pragma unroll
    for (int tj = 0; tj < 2; ++tj) {
      const v8f& acc = (ti == 0) ? (tj == 0 ? acc00 : acc01)
                                 : (tj == 0 ? acc10 : acc11);
#pragma unroll
      for (int r = 0; r < 8; ++r) {
        const int m = m0 + ti * 16 + r + 8 * h;   // C-layout: VGPR r -> row r+8h
        const int n = n0 + tj * 16 + nloc;        //           lane  -> col
        const float val = acc[r] + bias[n];
        if (mode == 0) {
          outF[(size_t)m * N + n] = val;
        } else {
          const int sec  = n >> 10;          // 0=q 1=k 2=v
          const int head = (n >> 6) & 15;
          const int d    = n & 63;
          const int b    = m >> 11;          // m = b*SEQ + t
          const int t    = m & (SEQ - 1);
          const size_t bh = (size_t)b * NHEADS + head;
          if (sec == 0) {
            qO[(bh * SEQ + t) * HEADD + d] = (__bf16)val;
          } else if (sec == 1) {
            kO[(bh * SEQ + t) * HEADD + d] = (__bf16)val;
          } else {
            // V stored transposed: [B,H,D,T]
            vO[(bh * HEADD + d) * SEQ + t] = (__bf16)val;
          }
        }
      }
    }
  }
}

// ---------------------------------------------------------------------------
// Flash attention forward: one block per (b, h, 64-row q tile), 4 waves,
// each wave owns 16 query rows. Keys streamed in 32-chunks with online
// softmax; no score matrix is ever materialized (avoids ~2 GB of traffic).
// V is pre-transposed [B,H,D,T] so every fragment load is contiguous b128.
// ---------------------------------------------------------------------------
__global__ __launch_bounds__(128)
void attn_fwd_wmma(const __bf16* __restrict__ Q, const __bf16* __restrict__ Kc,
                   const __bf16* __restrict__ Vt, __bf16* __restrict__ Oc) {
  __shared__ __align__(16) __bf16 pLds[4][16 * 32];  // per-wave P staging (1 KB)

  const int wid  = threadIdx.x >> 5;
  const int lane = threadIdx.x & 31;
  const int nloc = lane & 15;
  const int h    = lane >> 4;

  const int tile   = blockIdx.x & 31;   // T/64 = 32 q tiles
  const int bh     = blockIdx.x >> 5;   // b*NHEADS + head
  const int headId = bh & 15;
  const int bId    = bh >> 4;
  const size_t base = (size_t)bh * SEQ * HEADD;   // same size for [T,D] and [D,T]

  const int q0 = tile * 64 + wid * 16;  // this wave's first query row

  // Q fragments resident in VGPRs (D=64 -> two K=32 chunks)
  v16bf aq0 = load_frag_k_contig(Q + base + (size_t)q0 * HEADD,      HEADD);
  v16bf aq1 = load_frag_k_contig(Q + base + (size_t)q0 * HEADD + 32, HEADD);

  v8f o0 = {}, o1 = {}, o2 = {}, o3 = {};   // 16 x 64 output accumulator
  float mrow[8], lrow[8];
#pragma unroll
  for (int r = 0; r < 8; ++r) { mrow[r] = -1e30f; lrow[r] = 0.f; }
  const float scale = 0.125f;               // 1/sqrt(64)

  const int kend = tile * 64 + 64;          // causal: keys <= block's max query
  for (int kb = 0; kb < kend; kb += 32) {
    // S tiles: Q(16x64) @ K^T(64x32). K rows are contiguous-D -> contig loader.
    const __bf16* kp = Kc + base + (size_t)kb * HEADD;
    v16bf bk0a = load_frag_k_contig(kp,                  HEADD);
    v16bf bk0b = load_frag_k_contig(kp + 32,             HEADD);
    v16bf bk1a = load_frag_k_contig(kp + 16 * HEADD,     HEADD);
    v16bf bk1b = load_frag_k_contig(kp + 16 * HEADD + 32, HEADD);
    // Prefetch next key chunk of K.
    __builtin_prefetch(kp + 32 * HEADD, 0, 1);
    v8f s0 = {}, s1 = {};
    s0 = wmma_bf16(aq0, bk0a, s0);
    s0 = wmma_bf16(aq1, bk0b, s0);
    s1 = wmma_bf16(aq0, bk1a, s1);
    s1 = wmma_bf16(aq1, bk1b, s1);

    // Online softmax. C-layout: element r -> query row q0+r+8h; lane -> key col.
#pragma unroll
    for (int r = 0; r < 8; ++r) {
      const int qIdx = q0 + r + 8 * h;
      float f0 = s0[r] * scale;
      float f1 = s1[r] * scale;
      if (kb + nloc > qIdx)      f0 = -1e30f;   // causal mask
      if (kb + 16 + nloc > qIdx) f1 = -1e30f;
      float mx = fmaxf(f0, f1);
#pragma unroll
      for (int msk = 1; msk < 16; msk <<= 1)
        mx = fmaxf(mx, __shfl_xor(mx, msk, 32));   // reduce over 16-lane halves
      const float mnew = fmaxf(mrow[r], mx);
      const float p0 = __expf(f0 - mnew);
      const float p1 = __expf(f1 - mnew);
      float ps = p0 + p1;
#pragma unroll
      for (int msk = 1; msk < 16; msk <<= 1)
        ps += __shfl_xor(ps, msk, 32);
      const float cs = __expf(mrow[r] - mnew);
      lrow[r] = lrow[r] * cs + ps;
      mrow[r] = mnew;
      o0[r] *= cs; o1[r] *= cs; o2[r] *= cs; o3[r] *= cs;
      // Stage P (bf16) row-major 16x32 in this wave's LDS slab for A-frag reload.
      const int prow = r + 8 * h;
      pLds[wid][prow * 32 + nloc]      = (__bf16)p0;
      pLds[wid][prow * 32 + 16 + nloc] = (__bf16)p1;
    }
    __syncthreads();   // uniform trip count across all 4 waves in this block

    // O += P(16x32) @ V(32x64). V^T rows are d (lane), keys contiguous -> b128.
    v16bf ap  = load_frag_k_contig(&pLds[wid][0], 32);
    const __bf16* vp = Vt + base + kb;            // [D][T] layout, column kb
    v16bf bv0 = load_frag_k_contig(vp,            SEQ);
    v16bf bv1 = load_frag_k_contig(vp + 16 * SEQ, SEQ);
    v16bf bv2 = load_frag_k_contig(vp + 32 * SEQ, SEQ);
    v16bf bv3 = load_frag_k_contig(vp + 48 * SEQ, SEQ);
    o0 = wmma_bf16(ap, bv0, o0);
    o1 = wmma_bf16(ap, bv1, o1);
    o2 = wmma_bf16(ap, bv2, o2);
    o3 = wmma_bf16(ap, bv3, o3);
    __syncthreads();
  }

  // Normalize and store to [B,T,C] bf16 (heads re-interleaved for out-proj).
#pragma unroll
  for (int r = 0; r < 8; ++r) {
    const float inv = 1.0f / lrow[r];
    const int qIdx = q0 + r + 8 * h;
    const size_t rowbase =
        ((size_t)bId * SEQ + qIdx) * HIDDEN + (size_t)headId * HEADD;
    Oc[rowbase +  0 + nloc] = (__bf16)(o0[r] * inv);
    Oc[rowbase + 16 + nloc] = (__bf16)(o1[r] * inv);
    Oc[rowbase + 32 + nloc] = (__bf16)(o2[r] * inv);
    Oc[rowbase + 48 + nloc] = (__bf16)(o3[r] * inv);
  }
}

// ---------------------------------------------------------------------------
// Launch
// ---------------------------------------------------------------------------
extern "C" void kernel_launch(void* const* d_in, const int* in_sizes, int n_in,
                              void* d_out, int out_size, void* d_ws, size_t ws_size,
                              hipStream_t stream) {
  const float* x     = (const float*)d_in[0];
  const float* w_qkv = (const float*)d_in[1];
  const float* b_qkv = (const float*)d_in[2];
  const float* w_out = (const float*)d_in[3];
  const float* b_out = (const float*)d_in[4];
  float* out = (float*)d_out;

  char* ws = (char*)d_ws;
  size_t off = 0;
  auto carve = [&](size_t bytes) -> void* {
    void* p = ws + off;
    off += (bytes + 255) & ~(size_t)255;
    return p;
  };
  __bf16* xb    = (__bf16*)carve((size_t)MTOT * HIDDEN * 2);       // 8 MiB
  __bf16* wqkvT = (__bf16*)carve((size_t)3 * HIDDEN * HIDDEN * 2); // 6 MiB
  __bf16* woutT = (__bf16*)carve((size_t)HIDDEN * HIDDEN * 2);     // 2 MiB
  __bf16* qb    = (__bf16*)carve((size_t)MTOT * HIDDEN * 2);       // 8 MiB
  __bf16* kb    = (__bf16*)carve((size_t)MTOT * HIDDEN * 2);       // 8 MiB
  __bf16* vtb   = (__bf16*)carve((size_t)MTOT * HIDDEN * 2);       // 8 MiB ([B,H,D,T])
  __bf16* ab    = (__bf16*)carve((size_t)MTOT * HIDDEN * 2);       // 8 MiB

  {
    int n = MTOT * HIDDEN;
    cvt_f32_to_bf16<<<(n + 255) / 256, 256, 0, stream>>>(x, xb, n);
  }
  {
    int n = HIDDEN * 3 * HIDDEN;
    cvt_tr_f32_to_bf16<<<(n + 255) / 256, 256, 0, stream>>>(w_qkv, wqkvT, HIDDEN, 3 * HIDDEN);
  }
  {
    int n = HIDDEN * HIDDEN;
    cvt_tr_f32_to_bf16<<<(n + 255) / 256, 256, 0, stream>>>(w_out, woutT, HIDDEN, HIDDEN);
  }

  // QKV projection + bias, scatter into q/k [B,H,T,D] and v [B,H,D,T] bf16
  gemm_bf16_wmma<<<dim3(MTOT / 128, 3 * HIDDEN / 64), 256, 0, stream>>>(
      xb, wqkvT, b_qkv, nullptr, qb, kb, vtb, MTOT, 3 * HIDDEN, HIDDEN, 1);

  // Flash attention
  attn_fwd_wmma<<<BATCH * NHEADS * (SEQ / 64), 128, 0, stream>>>(qb, kb, vtb, ab);

  // Output projection + bias -> f32 result
  gemm_bf16_wmma<<<dim3(MTOT / 128, HIDDEN / 64), 256, 0, stream>>>(
      ab, woutT, b_out, out, nullptr, nullptr, nullptr, MTOT, HIDDEN, HIDDEN, 0);
}
